// CausalSelfAttention_19361712570429
// MI455X (gfx1250) — matrix operand order repaired and checked by reading, not compiled, not run on hardware
//
#include <hip/hip_runtime.h>

// ---------------------------------------------------------------------------
// Problem constants (B,T,C) = (4,2048,1024), 16 heads x 64 dim
// ---------------------------------------------------------------------------
#define BB   4
#define TT   2048
#define CC   1024
#define HH   16
#define DD   64
#define C3   3072

typedef __attribute__((ext_vector_type(16))) __bf16 v16bf;
typedef __attribute__((ext_vector_type(8)))  float  v8f;

union Frag {
    v16bf    v;
    unsigned u[8];
    uint4    q[2];
};

static __device__ __forceinline__ v8f vzero() {
    v8f z = {0.f,0.f,0.f,0.f,0.f,0.f,0.f,0.f};
    return z;
}

// float -> bf16 (round-to-nearest-even, NaN not special-cased)
static __device__ __forceinline__ unsigned short f2bf(float f) {
    unsigned u = __float_as_uint(f);
    u += 0x7FFFu + ((u >> 16) & 1u);
    return (unsigned short)(u >> 16);
}
static __device__ __forceinline__ unsigned pack2bf(float lo, float hi) {
    return (unsigned)f2bf(lo) | ((unsigned)f2bf(hi) << 16);
}
static __device__ __forceinline__ float bf2f(unsigned short u) {
    return __uint_as_float((unsigned)u << 16);
}

static __device__ __forceinline__ unsigned laneid() {
    return __builtin_amdgcn_mbcnt_lo(~0u, 0u);
}
static __device__ __forceinline__ float bperm(float x, int srcLane) {
    return __int_as_float(
        __builtin_amdgcn_ds_bpermute(srcLane << 2, __float_as_int(x)));
}

static __device__ __forceinline__ v8f wmma_bf16(const v16bf& a, const v16bf& b, v8f c) {
    return __builtin_amdgcn_wmma_f32_16x16x32_bf16(
        /*neg_a=*/false, a, /*neg_b=*/false, b,
        /*c_mod=*/(short)0, c, /*reuse_a=*/false, /*reuse_b=*/false);
}

// ---------------------------------------------------------------------------
// Kernel 1a: fp32 -> bf16 conversion (grid-stride)
// ---------------------------------------------------------------------------
__global__ void cvt_bf16_kernel(const float* __restrict__ src,
                                unsigned short* __restrict__ dst, int n) {
    for (int i = blockIdx.x * blockDim.x + threadIdx.x; i < n;
         i += gridDim.x * blockDim.x)
        dst[i] = f2bf(src[i]);
}

// ---------------------------------------------------------------------------
// Kernel 1b: fp32 [K,N] -> bf16 transposed [N,K] (weights, done once)
// ---------------------------------------------------------------------------
__global__ void cvt_bf16_t_kernel(const float* __restrict__ src,
                                  unsigned short* __restrict__ dst,
                                  int N, int K) {
    for (int i = blockIdx.x * blockDim.x + threadIdx.x; i < N * K;
         i += gridDim.x * blockDim.x) {
        const int n = i / K;
        const int k = i - n * K;
        dst[i] = f2bf(src[(size_t)k * N + n]);
    }
}

// ---------------------------------------------------------------------------
// Kernel 2: bf16 GEMM  C[M,N] = A[M,K] * Bt[N,K]^T, f32 accum.
// Bt is the N-major (pre-transposed) weight matrix -> all-contiguous loads.
// Block = 256 threads (8 waves), tile 128x128, wave tile 32x64 (2x4 WMMA).
// ---------------------------------------------------------------------------
template <bool BF16OUT>
__global__ void gemm_bf16_kernel(const unsigned short* __restrict__ A,
                                 const unsigned short* __restrict__ Bt,
                                 void* __restrict__ Cout,
                                 int M, int N, int K) {
    __shared__ __align__(16) unsigned short As[128 * 32];   // [row][k] 8KB
    __shared__ __align__(16) unsigned short Bs[128 * 32];   // [col][k] 8KB

    const int nblk = N >> 7;
    const int m0 = (blockIdx.x / nblk) << 7;
    const int n0 = (blockIdx.x % nblk) << 7;
    const int tid  = threadIdx.x;
    const int lane = tid & 31;
    const int w    = tid >> 5;
    const int wm   = w & 3;        // 4 waves along M (32 rows each)
    const int wn   = w >> 2;       // 2 waves along N (64 cols each)
    const int col  = lane & 15;
    const int hi   = lane >> 4;

    const int r    = tid & 127;    // staging row/col
    const int half = tid >> 7;     // staging k-half

    v8f acc[2][4];
#pragma unroll
    for (int i = 0; i < 2; ++i)
#pragma unroll
        for (int j = 0; j < 4; ++j) acc[i][j] = vzero();

    const unsigned short* aSrc = A  + (size_t)(m0 + r) * K + half * 16;
    const unsigned short* bSrc = Bt + (size_t)(n0 + r) * K + half * 16;
    uint4* aDst = (uint4*)(As + r * 32 + half * 16);
    uint4* bDst = (uint4*)(Bs + r * 32 + half * 16);

    for (int kk = 0; kk < K; kk += 32) {
        // --- stage A and B tiles: 2x b128 loads + 2x b128 LDS stores each ---
        {
            const uint4* sa = (const uint4*)(aSrc + kk);
            const uint4* sb = (const uint4*)(bSrc + kk);
            aDst[0] = sa[0];
            aDst[1] = sa[1];
            bDst[0] = sb[0];
            bDst[1] = sb[1];
            if (kk + 32 < K) {                        // global_prefetch_b8
                __builtin_prefetch(aSrc + kk + 32, 0, 1);
                __builtin_prefetch(bSrc + kk + 32, 0, 1);
            }
        }
        __syncthreads();

        Frag a[2], b[4];
#pragma unroll
        for (int i = 0; i < 2; ++i) {
            const unsigned short* base = As + (wm * 32 + i * 16 + col) * 32;
            a[i].q[0] = *(const uint4*)(base + hi * 16);        // K 16*hi + 0..7
            a[i].q[1] = *(const uint4*)(base + hi * 16 + 8);    // K 16*hi + 8..15
        }
#pragma unroll
        for (int j = 0; j < 4; ++j) {
            const unsigned short* base =
                Bs + (wn * 64 + j * 16 + col) * 32 + hi * 16;   // K 16*hi + 0..15
            b[j].q[0] = *(const uint4*)(base);
            b[j].q[1] = *(const uint4*)(base + 8);
        }
#pragma unroll
        for (int i = 0; i < 2; ++i)
#pragma unroll
            for (int j = 0; j < 4; ++j)
                acc[i][j] = wmma_bf16(a[i].v, b[j].v, acc[i][j]);
        __syncthreads();
    }

    // --- store: row = vgpr + 8*hi, col = lane&15 (branch-free) ---
#pragma unroll
    for (int i = 0; i < 2; ++i)
#pragma unroll
        for (int j = 0; j < 4; ++j) {
            const int rb = m0 + wm * 32 + i * 16 + 8 * hi;
            const int c  = n0 + wn * 64 + j * 16 + col;
#pragma unroll
            for (int v = 0; v < 8; ++v) {
                const size_t off = (size_t)(rb + v) * N + c;
                if constexpr (BF16OUT)
                    ((unsigned short*)Cout)[off] = f2bf(acc[i][j][v]);
                else
                    ((float*)Cout)[off] = acc[i][j][v];
            }
        }
}

// ---------------------------------------------------------------------------
// Kernel 3: RoPE in-place on q and k halves of qkv (bf16).
// ---------------------------------------------------------------------------
__global__ void rope_kernel(unsigned short* __restrict__ qkv) {
    int idx = blockIdx.x * blockDim.x + threadIdx.x;   // B*T*H*32 threads
    const int j = idx & 31;  idx >>= 5;
    const int h = idx & 15;  idx >>= 4;
    const int t = idx & 2047; idx >>= 11;
    const int b = idx;
    if (b >= BB) return;

    // inv_freq = 10000^(-j/32) = exp(-j * ln(10000)/32)
    const float ang = (float)t * __expf(-(float)j * 0.28782313662425576f);
    const float c = __cosf(ang), s = __sinf(ang);

    unsigned short* base = qkv + ((size_t)(b * TT + t)) * C3 + h * DD;
#pragma unroll
    for (int part = 0; part < 2; ++part) {           // q then k
        unsigned short* p = base + part * CC;
        const float x1 = bf2f(p[j]);
        const float x2 = bf2f(p[j + 32]);
        p[j]      = f2bf(x1 * c - x2 * s);
        p[j + 32] = f2bf(x2 * c + x1 * s);
    }
}

// ---------------------------------------------------------------------------
// Kernel 4: V transpose -> vt[b][h][d][t] (bf16) for contiguous PV B-frags.
// ---------------------------------------------------------------------------
__global__ void vtrans_kernel(const unsigned short* __restrict__ qkv,
                              unsigned short* __restrict__ vt) {
    int idx = blockIdx.x * blockDim.x + threadIdx.x;   // B*H*D*T threads
    const int t = idx & 2047; int r = idx >> 11;
    const int d = r & 63;     r >>= 6;
    const int h = r & 15;     r >>= 4;
    const int b = r;
    if (b >= BB) return;
    vt[(size_t)idx] =
        qkv[((size_t)(b * TT + t)) * C3 + 2 * CC + h * DD + d];
}

// ---------------------------------------------------------------------------
// Kernel 5: flash attention. 8 waves/block, one wave per 16-row query tile.
// S^T = K * Q^T  (so P -> A-fragment conversion is lane-local),
// O   = P * V    (V from transposed vt buffer, pipelined 1 frag deep).
// kb loop kept at unroll 1 to avoid scratch spills.
// ---------------------------------------------------------------------------
__global__ void __launch_bounds__(256, 1)
attn_kernel(const unsigned short* __restrict__ qkv,
            const unsigned short* __restrict__ vt,
            unsigned short* __restrict__ yb) {
    const int lane = threadIdx.x & 31;
    const int gw   = blockIdx.x * 8 + (threadIdx.x >> 5);
    const int qt   = gw & 127;            // T/16 tiles
    const int bh   = gw >> 7;
    const int b    = bh >> 4, h = bh & 15;
    const int Q0   = qt << 4;
    const int col  = lane & 15;
    const int hi   = lane >> 4;
    const int qa   = Q0 + col;            // this lane's query row (for stats)

    // ---- Q^T B-fragments (loop invariant): chunk c covers d = 32c..32c+31 ----
    Frag qf[2];
    {
        const unsigned short* qrow =
            qkv + ((size_t)(b * TT + Q0 + col)) * C3 + h * DD;
#pragma unroll
        for (int c = 0; c < 2; ++c) {
            const uint4* p = (const uint4*)(qrow + c * 32);
            qf[c].q[0] = p[hi];
            qf[c].q[1] = p[2 + hi];
        }
    }

    v8f o[4];
#pragma unroll
    for (int nc = 0; nc < 4; ++nc) o[nc] = vzero();
    float mrow = -3.0e38f, lrow = 0.0f;

    // base pointers (lane-fixed parts)
    const unsigned short* kbase =
        qkv + (size_t)b * TT * C3 + CC + h * DD;             // + t*C3
    const unsigned short* vbase =
        vt + ((size_t)((b * HH + h) * DD + col)) * TT + hi * 8;   // + nc*16*TT + kb

#pragma unroll 1
    for (int kb = 0; kb < Q0 + 16; kb += 32) {
        // ---- hoisted K loads (4 fragments) ----
        Frag ka00, ka01, ka10, ka11;
        {
            const unsigned short* kr0 = kbase + (size_t)(kb + col) * C3;
            const unsigned short* kr1 = kr0 + (size_t)16 * C3;
            ka00.q[0] = *(const uint4*)(kr0 + hi * 8);
            ka00.q[1] = *(const uint4*)(kr0 + 16 + hi * 8);
            ka01.q[0] = *(const uint4*)(kr0 + 32 + hi * 8);
            ka01.q[1] = *(const uint4*)(kr0 + 48 + hi * 8);
            ka10.q[0] = *(const uint4*)(kr1 + hi * 8);
            ka10.q[1] = *(const uint4*)(kr1 + 16 + hi * 8);
            ka11.q[0] = *(const uint4*)(kr1 + 32 + hi * 8);
            ka11.q[1] = *(const uint4*)(kr1 + 48 + hi * 8);
            if (kb + 32 < Q0 + 16) {       // prefetch next iteration's K rows
                __builtin_prefetch(kr0 + (size_t)32 * C3, 0, 1);
                __builtin_prefetch(kr1 + (size_t)32 * C3, 0, 1);
            }
        }

        // ---- S^T = K * Q^T ----
        v8f st[2];
        st[0] = wmma_bf16(ka00.v, qf[0].v, vzero());
        st[0] = wmma_bf16(ka01.v, qf[1].v, st[0]);
        st[1] = wmma_bf16(ka10.v, qf[0].v, vzero());
        st[1] = wmma_bf16(ka11.v, qf[1].v, st[1]);

        // ---- scale + causal mask + row max ----
        // key(sub=0,j) = kb + 8*hi + j  -> mask iff j > thr0 = qa - kb - 8*hi
        // key(sub=1,j) = thr0 - 16 threshold
        const int thr0 = qa - kb - 8 * hi;
        const int thr1 = thr0 - 16;
        float mloc = -3.0e38f;
#pragma unroll
        for (int j = 0; j < 8; ++j) {
            float s0 = st[0][j] * 0.125f;
            float s1 = st[1][j] * 0.125f;
            if (j > thr0) s0 = -3.0e38f;
            if (j > thr1) s1 = -3.0e38f;
            st[0][j] = s0;
            st[1][j] = s1;
            mloc = fmaxf(mloc, fmaxf(s0, s1));
        }
        mloc = fmaxf(mloc, bperm(mloc, (int)(laneid() ^ 16)));
        const float mnew  = fmaxf(mrow, mloc);
        const float alpha = __expf(mrow - mnew);
        mrow = mnew;

        // ---- P = exp(S - m): build A-fragment lane-locally ----
        Frag pf;
        float ls = 0.0f;
#pragma unroll
        for (int i = 0; i < 4; ++i) {
            const float e0 = __expf(st[0][2 * i]     - mnew);
            const float e1 = __expf(st[0][2 * i + 1] - mnew);
            const float f0 = __expf(st[1][2 * i]     - mnew);
            const float f1 = __expf(st[1][2 * i + 1] - mnew);
            ls += e0 + e1 + f0 + f1;
            pf.u[i]     = pack2bf(e0, e1);   // keys 8*hi + 2i,2i+1
            pf.u[4 + i] = pack2bf(f0, f1);   // keys 16 + 8*hi + 2i,2i+1
        }
        lrow = lrow * alpha + ls + bperm(ls, (int)(laneid() ^ 16));

        // ---- rescale O by per-row alpha (row = vgpr + 8*hi) ----
        float av[8];
#pragma unroll
        for (int i = 0; i < 8; ++i) av[i] = bperm(alpha, i + 8 * hi);
#pragma unroll
        for (int nc = 0; nc < 4; ++nc)
#pragma unroll
            for (int v = 0; v < 8; ++v) o[nc][v] *= av[v];

        // ---- O += P * V, V loads pipelined one fragment ahead ----
        const unsigned short* vr = vbase + kb;
        Frag vcur, vnext;
        vcur.q[0] = *(const uint4*)(vr);
        vcur.q[1] = *(const uint4*)(vr + 16);
#pragma unroll
        for (int nc = 0; nc < 4; ++nc) {
            if (nc < 3) {
                const unsigned short* vn = vr + (size_t)((nc + 1) * 16) * TT;
                vnext.q[0] = *(const uint4*)(vn);
                vnext.q[1] = *(const uint4*)(vn + 16);
            }
            o[nc] = wmma_bf16(pf.v, vcur.v, o[nc]);
            vcur = vnext;
        }
    }

    // ---- normalize and store y (bf16) ----
    const float inv = 1.0f / lrow;
    float nv[8];
#pragma unroll
    for (int i = 0; i < 8; ++i) nv[i] = bperm(inv, i + 8 * hi);
#pragma unroll
    for (int nc = 0; nc < 4; ++nc)
#pragma unroll
        for (int v = 0; v < 8; ++v) {
            const int row = Q0 + v + 8 * hi;
            yb[((size_t)(b * TT + row)) * CC + h * DD + nc * 16 + col] =
                f2bf(o[nc][v] * nv[v]);
        }
}

// ---------------------------------------------------------------------------
// Host-side launcher
// ---------------------------------------------------------------------------
extern "C" void kernel_launch(void* const* d_in, const int* in_sizes, int n_in,
                              void* d_out, int out_size, void* d_ws, size_t ws_size,
                              hipStream_t stream) {
    (void)in_sizes; (void)n_in; (void)out_size; (void)ws_size;
    const float* x  = (const float*)d_in[0];   // [B,T,C]
    const float* Wa = (const float*)d_in[1];   // [C,3C]
    const float* Wp = (const float*)d_in[2];   // [C,C]
    float* out = (float*)d_out;                // [B,T,C] fp32

    char* ws = (char*)d_ws;
    size_t off = 0;
    auto take = [&](size_t bytes) {
        char* p = ws + off;
        off = (off + bytes + 255) & ~(size_t)255;
        return p;
    };
    unsigned short* xb   = (unsigned short*)take((size_t)BB * TT * CC * 2);  // x bf16
    unsigned short* wat  = (unsigned short*)take((size_t)CC * C3 * 2);       // W_attn^T bf16 [3C][C]
    unsigned short* wpt  = (unsigned short*)take((size_t)CC * CC * 2);       // W_proj^T bf16 [C][C]
    unsigned short* qkvb = (unsigned short*)take((size_t)BB * TT * C3 * 2);  // qkv bf16
    unsigned short* vtb  = (unsigned short*)take((size_t)BB * TT * CC * 2);  // V^T bf16
    unsigned short* ybuf = (unsigned short*)take((size_t)BB * TT * CC * 2);  // attn out bf16

    const int nx = BB * TT * CC;    // 8388608

    cvt_bf16_kernel<<<4096, 256, 0, stream>>>(x, xb, nx);
    cvt_bf16_t_kernel<<<4096, 256, 0, stream>>>(Wa, wat, C3, CC);  // [C,3C]->[3C,C]
    cvt_bf16_t_kernel<<<2048, 256, 0, stream>>>(Wp, wpt, CC, CC);  // [C,C]->[C,C]

    // qkv = x @ W_attn : M=8192, N=3072, K=1024 -> bf16 out
    gemm_bf16_kernel<true><<<(8192 / 128) * (3072 / 128), 256, 0, stream>>>(
        xb, wat, (void*)qkvb, 8192, 3072, 1024);

    rope_kernel<<<(BB * TT * HH * 32) / 256, 256, 0, stream>>>(qkvb);
    vtrans_kernel<<<(BB * HH * DD * TT) / 256, 256, 0, stream>>>(qkvb, vtb);

    // attention: B*H*(T/16) = 8192 waves, 8 waves/block
    attn_kernel<<<1024, 256, 0, stream>>>(qkvb, vtb, ybuf);

    // out = y @ W_proj : M=8192, N=1024, K=1024 -> fp32 out
    gemm_bf16_kernel<false><<<(8192 / 128) * (1024 / 128), 256, 0, stream>>>(
        ybuf, wpt, (void*)out, 8192, 1024, 1024);
}
